// MultiHeadAttention_69973607187164
// MI455X (gfx1250) — matrix operand order repaired and checked by reading, not compiled, not run on hardware
//
#include <hip/hip_runtime.h>

// ---------------------------------------------------------------------------
// MI455X / gfx1250 multi-head attention forward.
// GEMMs on v_wmma_f32_16x16x32_bf16; bf16-sourced GEMMs stage tiles with
// global_load_async_to_lds_b128 (ASYNCcnt), fp32-sourced GEMMs use a
// register-staged software pipeline (convert fp32->bf16 in flight).
// ---------------------------------------------------------------------------

typedef __attribute__((ext_vector_type(16))) __bf16       v16bf;
typedef __attribute__((ext_vector_type(8)))  float        v8f;
typedef __attribute__((ext_vector_type(4)))  unsigned int u32x4;

#define WMMA_BF16(a, b, c) \
  __builtin_amdgcn_wmma_f32_16x16x32_bf16(false, (a), false, (b), (short)0, (c), false, false)

// Async copy 16 bytes global -> LDS (per-lane), tracked by ASYNCcnt.
#define ASYNC_G2L_B128(ldsOff, gptr)                                       \
  asm volatile("global_load_async_to_lds_b128 %0, %1, off"                 \
               :: "v"(ldsOff), "v"(gptr) : "memory")
#define WAIT_ASYNC0() asm volatile("s_wait_asynccnt 0" ::: "memory")

// Problem constants (B=8, L=512, D=1024, H=16, d_k=64)
#define BQ   8
#define LQ   512
#define DM   1024
#define NH   16
#define DK   64
#define MROWS (BQ * LQ)  // 4096

// ---------------------------------------------------------------------------
// WMMA fragment loaders (wave32; contiguous k order k = 16*(lane/16) + i, same in A and B)
// ---------------------------------------------------------------------------
// A fragment 16x32: lanes 0-15 M=lane K=0..15; lanes 16-31 K=16..31
__device__ __forceinline__ v16bf load_a_frag(const __bf16* t, int m0, int k0,
                                             int stride, int lane) {
  const int half = lane >> 4;
  const __bf16* row = t + (m0 + (lane & 15)) * stride + k0 + half * 16;
  v16bf a;
#pragma unroll
  for (int i = 0; i < 16; ++i) a[i] = row[i];
  return a;
}

// B fragment for C = A @ B^T, tile stored Bt[n][k] row-major.
__device__ __forceinline__ v16bf load_bt_frag(const __bf16* t, int n0, int k0,
                                              int stride, int lane) {
  const int half = lane >> 4;
  const __bf16* row = t + (n0 + (lane & 15)) * stride + k0 + half * 16;
  v16bf b;
#pragma unroll
  for (int i = 0; i < 16; ++i) b[i] = row[i];
  return b;
}

// B fragment for C = A @ B, tile stored Bt[k][n] row-major.
__device__ __forceinline__ v16bf load_bn_frag(const __bf16* t, int n0,
                                              int stride, int lane) {
  const int half = lane >> 4;
  const int n = n0 + (lane & 15);
  v16bf b;
#pragma unroll
  for (int i = 0; i < 16; ++i) b[i] = t[(half * 16 + i) * stride + n];
  return b;
}

__device__ __forceinline__ v8f zero_v8f() {
  v8f z = {0.f, 0.f, 0.f, 0.f, 0.f, 0.f, 0.f, 0.f};
  return z;
}

// ---------------------------------------------------------------------------
// Kernel 1: QKV projection (X fp32 @ W^T + bias) -> head-split bf16.
// Block 128x128, K-step 32, 8 waves (4M x 2N), wave tile 32x64 (2x4 acc).
// Software pipeline: next tile's global loads issued between LDS-ready
// barrier and the WMMA block.
// ---------------------------------------------------------------------------
__global__ __launch_bounds__(256)
void proj_gemm_kernel(const float* __restrict__ Xin, const float* __restrict__ W,
                      const float* __restrict__ bias, __bf16* __restrict__ out) {
  __shared__ __attribute__((aligned(16))) __bf16 As[128 * 40];
  __shared__ __attribute__((aligned(16))) __bf16 Bs[128 * 40];
  const int tid  = threadIdx.x;
  const int lane = tid & 31, wave = tid >> 5;
  const int gm0 = blockIdx.x * 128, gn0 = blockIdx.y * 128;
  const int wm = (wave >> 1) * 32, wn = (wave & 1) * 64;

  v8f acc[2][4];
#pragma unroll
  for (int i = 0; i < 2; ++i)
#pragma unroll
    for (int j = 0; j < 4; ++j) acc[i][j] = zero_v8f();

  float4 ar[4], br[4];
#pragma unroll
  for (int it = 0; it < 4; ++it) {  // prologue: tile k0=0 into registers
    const int f = tid + it * 256;
    const int r = f >> 3, c = (f & 7) << 2;
    ar[it] = *(const float4*)(Xin + (size_t)(gm0 + r) * DM + c);
    br[it] = *(const float4*)(W + (size_t)(gn0 + r) * DM + c);
  }

  for (int k0 = 0; k0 < DM; k0 += 32) {
    __syncthreads();  // previous iteration's LDS reads complete
#pragma unroll
    for (int it = 0; it < 4; ++it) {  // stage regs -> LDS (fp32 -> bf16)
      const int f = tid + it * 256;
      const int r = f >> 3, c = (f & 7) << 2;
      __bf16* da = As + r * 40 + c;
      da[0] = (__bf16)ar[it].x; da[1] = (__bf16)ar[it].y;
      da[2] = (__bf16)ar[it].z; da[3] = (__bf16)ar[it].w;
      __bf16* db = Bs + r * 40 + c;
      db[0] = (__bf16)br[it].x; db[1] = (__bf16)br[it].y;
      db[2] = (__bf16)br[it].z; db[3] = (__bf16)br[it].w;
    }
    __syncthreads();
    if (k0 + 32 < DM) {  // prefetch next tile (overlaps with WMMA below)
#pragma unroll
      for (int it = 0; it < 4; ++it) {
        const int f = tid + it * 256;
        const int r = f >> 3, c = (f & 7) << 2;
        ar[it] = *(const float4*)(Xin + (size_t)(gm0 + r) * DM + k0 + 32 + c);
        br[it] = *(const float4*)(W + (size_t)(gn0 + r) * DM + k0 + 32 + c);
      }
    }
    const v16bf a0 = load_a_frag(As, wm, 0, 40, lane);
    const v16bf a1 = load_a_frag(As, wm + 16, 0, 40, lane);
#pragma unroll
    for (int j = 0; j < 4; ++j) {
      const v16bf b = load_bt_frag(Bs, wn + j * 16, 0, 40, lane);
      acc[0][j] = WMMA_BF16(a0, b, acc[0][j]);
      acc[1][j] = WMMA_BF16(a1, b, acc[1][j]);
    }
  }

  const int nl = lane & 15, hl = lane >> 4;
#pragma unroll
  for (int mi = 0; mi < 2; ++mi)
#pragma unroll
    for (int ni = 0; ni < 4; ++ni) {
      const int gn = gn0 + wn + ni * 16 + nl;
      const int h = gn >> 6, dk = gn & 63;
      const float bv = bias[gn];
#pragma unroll
      for (int r = 0; r < 8; ++r) {
        const int gm = gm0 + wm + mi * 16 + hl * 8 + r;
        const int bb = gm >> 9, l = gm & 511;
        out[(((size_t)(bb * NH + h) * LQ) + l) * DK + dk] = (__bf16)(acc[mi][ni][r] + bv);
      }
    }
}

// ---------------------------------------------------------------------------
// Kernel 2: scores = Qh @ Kh^T * 0.125 per (b,h) -> attns region of d_out.
// Block 128x128; full K=64 staged once via async-to-LDS, then 2 WMMA K-steps
// with no further barriers.
// ---------------------------------------------------------------------------
__global__ __launch_bounds__(256)
void scores_gemm_kernel(const __bf16* __restrict__ qh, const __bf16* __restrict__ kh,
                        float* __restrict__ attn) {
  __shared__ __attribute__((aligned(16))) __bf16 As[128 * 72];  // 128 x 64(+8)
  __shared__ __attribute__((aligned(16))) __bf16 Bs[128 * 72];
  const int tid  = threadIdx.x;
  const int lane = tid & 31, wave = tid >> 5;
  const int gm0 = blockIdx.x * 128, gn0 = blockIdx.y * 128;
  const int z = blockIdx.z;  // b*16+h
  const int b = z >> 4, h = z & 15;
  const __bf16* Qb = qh + (size_t)z * LQ * DK;
  const __bf16* Kb = kh + (size_t)z * LQ * DK;
  const int wm = (wave >> 1) * 32, wn = (wave & 1) * 64;

  const unsigned asb = (unsigned)(uintptr_t)&As[0];
  const unsigned bsb = (unsigned)(uintptr_t)&Bs[0];
#pragma unroll
  for (int it = 0; it < 4; ++it) {  // whole 128x64 tiles, async to LDS
    const int f = tid + it * 256;
    const int r = f >> 3, c = (f & 7) << 3;
    const unsigned lo = (unsigned)((r * 72 + c) * 2);
    ASYNC_G2L_B128(asb + lo, Qb + (size_t)(gm0 + r) * DK + c);
    ASYNC_G2L_B128(bsb + lo, Kb + (size_t)(gn0 + r) * DK + c);
  }
  WAIT_ASYNC0();
  __syncthreads();

  v8f acc[2][4];
#pragma unroll
  for (int i = 0; i < 2; ++i)
#pragma unroll
    for (int j = 0; j < 4; ++j) acc[i][j] = zero_v8f();

#pragma unroll
  for (int kk = 0; kk < DK; kk += 32) {
    const v16bf a0 = load_a_frag(As, wm, kk, 72, lane);
    const v16bf a1 = load_a_frag(As, wm + 16, kk, 72, lane);
#pragma unroll
    for (int j = 0; j < 4; ++j) {
      const v16bf b2 = load_bt_frag(Bs, wn + j * 16, kk, 72, lane);
      acc[0][j] = WMMA_BF16(a0, b2, acc[0][j]);
      acc[1][j] = WMMA_BF16(a1, b2, acc[1][j]);
    }
  }

  float* Ab = attn + ((size_t)(h * BQ + b) * LQ) * LQ;
  const int nl = lane & 15, hl = lane >> 4;
#pragma unroll
  for (int mi = 0; mi < 2; ++mi)
#pragma unroll
    for (int ni = 0; ni < 4; ++ni) {
      const int gn = gn0 + wn + ni * 16 + nl;
#pragma unroll
      for (int r = 0; r < 8; ++r) {
        const int gm = gm0 + wm + mi * 16 + hl * 8 + r;
        Ab[(size_t)gm * LQ + gn] = 0.125f * acc[mi][ni][r];  // 1/sqrt(64)
      }
    }
}

// ---------------------------------------------------------------------------
// Kernel 3: row softmax over 512 cols; one wave32 per row; shfl_xor trees.
// ---------------------------------------------------------------------------
__global__ __launch_bounds__(256)
void softmax_kernel(float* __restrict__ attn, __bf16* __restrict__ P) {
  const int lane = threadIdx.x & 31, wave = threadIdx.x >> 5;
  const size_t row = (size_t)blockIdx.x * 8 + wave;
  float* a = attn + row * LQ;
  __bf16* p = P + row * LQ;
  float v[16];
  float mx = -3.4e38f;
#pragma unroll
  for (int i = 0; i < 16; ++i) { v[i] = a[lane + 32 * i]; mx = fmaxf(mx, v[i]); }
#pragma unroll
  for (int off = 16; off > 0; off >>= 1) mx = fmaxf(mx, __shfl_xor(mx, off, 32));
  float s = 0.f;
#pragma unroll
  for (int i = 0; i < 16; ++i) { v[i] = __expf(v[i] - mx); s += v[i]; }
#pragma unroll
  for (int off = 16; off > 0; off >>= 1) s += __shfl_xor(s, off, 32);
  const float inv = 1.0f / s;
#pragma unroll
  for (int i = 0; i < 16; ++i) {
    const float pr = v[i] * inv;
    a[lane + 32 * i] = pr;
    p[lane + 32 * i] = (__bf16)pr;
  }
}

// ---------------------------------------------------------------------------
// Kernel 4: O = P @ Vh per (b,h).  M=512 (block 256), N=64, K=512.
// Double-buffered async-to-LDS: batch for step s+1 issued while WMMAs of
// step s run; one s_wait_asynccnt per step.
// ---------------------------------------------------------------------------
__global__ __launch_bounds__(256)
void pv_gemm_kernel(const __bf16* __restrict__ P, const __bf16* __restrict__ vh,
                    __bf16* __restrict__ O) {
  __shared__ __attribute__((aligned(16))) __bf16 As[2][256 * 40];  // 256x32
  __shared__ __attribute__((aligned(16))) __bf16 Bs[2][32 * 72];   // 32x64 [k][n]
  const int tid  = threadIdx.x;
  const int lane = tid & 31, wave = tid >> 5;
  const int gm0 = blockIdx.x * 256;
  const int z = blockIdx.z;  // b*16+h
  const int b = z >> 4, h = z & 15;
  const __bf16* Pb = P + ((size_t)(h * BQ + b) * LQ) * LQ;
  const __bf16* Vb = vh + (size_t)z * LQ * DK;
  const int wm = wave * 32;

  const unsigned asb = (unsigned)(uintptr_t)&As[0][0];
  const unsigned bsb = (unsigned)(uintptr_t)&Bs[0][0];
  const int ra = tid >> 2, ca = (tid & 3) << 3;   // A chunk coords (+64 rows/it)
  const int rb = tid >> 3, cb = (tid & 7) << 3;   // B chunk coords

  // issue one K-step batch into buffer `buf`
  auto issue = [&](int buf, int k0) {
    const unsigned ao = asb + (unsigned)buf * (256 * 40 * 2);
    const unsigned bo = bsb + (unsigned)buf * (32 * 72 * 2);
#pragma unroll
    for (int it = 0; it < 4; ++it) {
      const int r = ra + it * 64;
      ASYNC_G2L_B128(ao + (unsigned)((r * 40 + ca) * 2),
                     Pb + (size_t)(gm0 + r) * LQ + k0 + ca);
    }
    ASYNC_G2L_B128(bo + (unsigned)((rb * 72 + cb) * 2),
                   Vb + (size_t)(k0 + rb) * DK + cb);
  };

  v8f acc[2][4];
#pragma unroll
  for (int i = 0; i < 2; ++i)
#pragma unroll
    for (int j = 0; j < 4; ++j) acc[i][j] = zero_v8f();

  issue(0, 0);
  for (int s = 0; s < 16; ++s) {
    const int cur = s & 1;
    WAIT_ASYNC0();
    __syncthreads();  // LDS[cur] visible to all waves
    if (s < 15) issue(1 - cur, (s + 1) * 32);  // overlaps with WMMAs below
    const v16bf a0 = load_a_frag(As[cur], wm, 0, 40, lane);
    const v16bf a1 = load_a_frag(As[cur], wm + 16, 0, 40, lane);
#pragma unroll
    for (int j = 0; j < 4; ++j) {
      const v16bf b2 = load_bn_frag(Bs[cur], j * 16, 72, lane);
      acc[0][j] = WMMA_BF16(a0, b2, acc[0][j]);
      acc[1][j] = WMMA_BF16(a1, b2, acc[1][j]);
    }
    __syncthreads();  // all reads of LDS[cur] done before it is refilled
  }

  const int nl = lane & 15, hl = lane >> 4;
#pragma unroll
  for (int mi = 0; mi < 2; ++mi)
#pragma unroll
    for (int ni = 0; ni < 4; ++ni) {
      const int gn = ni * 16 + nl;  // dk within head
#pragma unroll
      for (int r = 0; r < 8; ++r) {
        const int gm = gm0 + wm + mi * 16 + hl * 8 + r;  // q index
        O[((size_t)(b * LQ + gm)) * DM + h * DK + gn] = (__bf16)acc[mi][ni][r];
      }
    }
}

// ---------------------------------------------------------------------------
// Kernel 5: X = O @ Wf^T + bf + residual(q) -> fp32 pre-LayerNorm.
// Same pipelined 128x128 structure as proj; A is bf16, B fp32->bf16.
// ---------------------------------------------------------------------------
__global__ __launch_bounds__(256)
void final_gemm_kernel(const __bf16* __restrict__ O, const float* __restrict__ W,
                       const float* __restrict__ bias, const float* __restrict__ resid,
                       float* __restrict__ Xout) {
  __shared__ __attribute__((aligned(16))) __bf16 As[128 * 40];
  __shared__ __attribute__((aligned(16))) __bf16 Bs[128 * 40];
  const int tid  = threadIdx.x;
  const int lane = tid & 31, wave = tid >> 5;
  const int gm0 = blockIdx.x * 128, gn0 = blockIdx.y * 128;
  const int wm = (wave >> 1) * 32, wn = (wave & 1) * 64;

  v8f acc[2][4];
#pragma unroll
  for (int i = 0; i < 2; ++i)
#pragma unroll
    for (int j = 0; j < 4; ++j) acc[i][j] = zero_v8f();

  u32x4 ar[2];
  float4 br[4];
#pragma unroll
  for (int it = 0; it < 2; ++it) {  // A: 128x32 bf16 = 512 chunks, 2/thread
    const int f = tid + it * 256;
    const int r = f >> 2, c = (f & 3) << 3;
    ar[it] = *(const u32x4*)(O + (size_t)(gm0 + r) * DM + c);
  }
#pragma unroll
  for (int it = 0; it < 4; ++it) {  // B: 128x32 fp32
    const int f = tid + it * 256;
    const int r = f >> 3, c = (f & 7) << 2;
    br[it] = *(const float4*)(W + (size_t)(gn0 + r) * DM + c);
  }

  for (int k0 = 0; k0 < DM; k0 += 32) {
    __syncthreads();
#pragma unroll
    for (int it = 0; it < 2; ++it) {
      const int f = tid + it * 256;
      const int r = f >> 2, c = (f & 3) << 3;
      *(u32x4*)(As + r * 40 + c) = ar[it];
    }
#pragma unroll
    for (int it = 0; it < 4; ++it) {
      const int f = tid + it * 256;
      const int r = f >> 3, c = (f & 7) << 2;
      __bf16* db = Bs + r * 40 + c;
      db[0] = (__bf16)br[it].x; db[1] = (__bf16)br[it].y;
      db[2] = (__bf16)br[it].z; db[3] = (__bf16)br[it].w;
    }
    __syncthreads();
    if (k0 + 32 < DM) {
#pragma unroll
      for (int it = 0; it < 2; ++it) {
        const int f = tid + it * 256;
        const int r = f >> 2, c = (f & 3) << 3;
        ar[it] = *(const u32x4*)(O + (size_t)(gm0 + r) * DM + k0 + 32 + c);
      }
#pragma unroll
      for (int it = 0; it < 4; ++it) {
        const int f = tid + it * 256;
        const int r = f >> 3, c = (f & 7) << 2;
        br[it] = *(const float4*)(W + (size_t)(gn0 + r) * DM + k0 + 32 + c);
      }
    }
    const v16bf a0 = load_a_frag(As, wm, 0, 40, lane);
    const v16bf a1 = load_a_frag(As, wm + 16, 0, 40, lane);
#pragma unroll
    for (int j = 0; j < 4; ++j) {
      const v16bf b2 = load_bt_frag(Bs, wn + j * 16, 0, 40, lane);
      acc[0][j] = WMMA_BF16(a0, b2, acc[0][j]);
      acc[1][j] = WMMA_BF16(a1, b2, acc[1][j]);
    }
  }

  const int nl = lane & 15, hl = lane >> 4;
#pragma unroll
  for (int mi = 0; mi < 2; ++mi)
#pragma unroll
    for (int ni = 0; ni < 4; ++ni) {
      const int gn = gn0 + wn + ni * 16 + nl;
      const float bv = bias[gn];
#pragma unroll
      for (int r = 0; r < 8; ++r) {
        const int gm = gm0 + wm + mi * 16 + hl * 8 + r;
        Xout[(size_t)gm * DM + gn] = acc[mi][ni][r] + bv + resid[(size_t)gm * DM + gn];
      }
    }
}

// ---------------------------------------------------------------------------
// Kernel 6: LayerNorm per row of 1024.
// ---------------------------------------------------------------------------
__global__ __launch_bounds__(256)
void layernorm_kernel(const float* __restrict__ Xin, const float* __restrict__ gamma,
                      const float* __restrict__ beta, float* __restrict__ out) {
  const size_t row = blockIdx.x;
  const float* x = Xin + row * DM;
  float* o = out + row * DM;
  const int tid = threadIdx.x, lane = tid & 31, wave = tid >> 5;
  float v[4];
  float s = 0.f, s2 = 0.f;
#pragma unroll
  for (int i = 0; i < 4; ++i) {
    v[i] = x[tid + 256 * i];
    s += v[i];
    s2 += v[i] * v[i];
  }
#pragma unroll
  for (int off = 16; off > 0; off >>= 1) {
    s  += __shfl_xor(s, off, 32);
    s2 += __shfl_xor(s2, off, 32);
  }
  __shared__ float r1[8], r2[8];
  if (lane == 0) { r1[wave] = s; r2[wave] = s2; }
  __syncthreads();
  float ts = 0.f, ts2 = 0.f;
#pragma unroll
  for (int i = 0; i < 8; ++i) { ts += r1[i]; ts2 += r2[i]; }
  const float mu = ts * (1.0f / 1024.0f);
  const float var = ts2 * (1.0f / 1024.0f) - mu * mu;
  const float rs = rsqrtf(var + 1e-5f);
#pragma unroll
  for (int i = 0; i < 4; ++i) {
    const int c = tid + 256 * i;
    o[c] = (v[i] - mu) * rs * gamma[c] + beta[c];
  }
}

// ---------------------------------------------------------------------------
// Launch
// ---------------------------------------------------------------------------
extern "C" void kernel_launch(void* const* d_in, const int* in_sizes, int n_in,
                              void* d_out, int out_size, void* d_ws, size_t ws_size,
                              hipStream_t stream) {
  (void)in_sizes; (void)n_in; (void)out_size; (void)ws_size;
  const float* q     = (const float*)d_in[0];
  const float* k     = (const float*)d_in[1];
  const float* v     = (const float*)d_in[2];
  const float* Wq    = (const float*)d_in[3];
  const float* bq    = (const float*)d_in[4];
  const float* Wk    = (const float*)d_in[5];
  const float* bk    = (const float*)d_in[6];
  const float* Wv    = (const float*)d_in[7];
  const float* bv    = (const float*)d_in[8];
  const float* Wf    = (const float*)d_in[9];
  const float* bf_   = (const float*)d_in[10];
  const float* gamma = (const float*)d_in[11];
  const float* beta  = (const float*)d_in[12];

  // Workspace layout (bytes)
  char* ws = (char*)d_ws;
  __bf16* qh = (__bf16*)(ws + 0);          //  8 MiB: [(b,h), l, dk] bf16
  __bf16* kh = (__bf16*)(ws + 8388608);    //  8 MiB
  __bf16* vh = (__bf16*)(ws + 16777216);   //  8 MiB
  __bf16* P  = (__bf16*)(ws + 25165824);   // 64 MiB: probs bf16, head-major rows
  __bf16* O  = (__bf16*)(ws + 92274688);   //  8 MiB: concat-head attn output bf16
  float*  X  = (float*)(ws + 100663296);   // 16 MiB: pre-LayerNorm fp32

  float* outputs = (float*)d_out;                       // [4096 x 1024]
  float* attns   = (float*)d_out + (size_t)MROWS * DM;  // [128 x 512 x 512]

  const dim3 gproj(MROWS / 128, DM / 128, 1);
  proj_gemm_kernel<<<gproj, 256, 0, stream>>>(q, Wq, bq, qh);
  proj_gemm_kernel<<<gproj, 256, 0, stream>>>(k, Wk, bk, kh);
  proj_gemm_kernel<<<gproj, 256, 0, stream>>>(v, Wv, bv, vh);

  const dim3 gsc(LQ / 128, LQ / 128, BQ * NH);
  scores_gemm_kernel<<<gsc, 256, 0, stream>>>(qh, kh, attns);

  softmax_kernel<<<(BQ * NH * LQ) / 8, 256, 0, stream>>>(attns, P);

  const dim3 gpv(LQ / 256, 1, BQ * NH);
  pv_gemm_kernel<<<gpv, 256, 0, stream>>>(P, vh, O);

  final_gemm_kernel<<<gproj, 256, 0, stream>>>(O, Wf, bf_, q, X);

  layernorm_kernel<<<MROWS, 256, 0, stream>>>(X, gamma, beta, outputs);
}
